// StateRNN_4904852652795
// MI455X (gfx1250) — matrix operand order repaired and checked by reading, not compiled, not run on hardware
//
#include <hip/hip_runtime.h>

#define HID   512
#define TLEN  512
#define DIN   6
#define KT1   17      // layer1 K padded 518 -> 544
#define KT2   32      // layer2 K = 1024
#define A1STR 544
#define A2STR 1024

typedef __bf16 bf16_t;
typedef __attribute__((ext_vector_type(16))) __bf16 v16bf;
typedef __attribute__((ext_vector_type(8)))  __bf16 v8bf;
typedef __attribute__((ext_vector_type(8)))  float  v8f;

__device__ __forceinline__ bf16_t f2bf(float f) {
  union { float f; unsigned u; } v; v.f = f;
  unsigned r = v.u + 0x7FFFu + ((v.u >> 16) & 1u);   // RNE
  unsigned short h = (unsigned short)(r >> 16);
  return __builtin_bit_cast(bf16_t, h);
}
__device__ __forceinline__ float sigf(float x) { return 1.0f / (1.0f + __expf(-x)); }

// A-fragment (16x32 bf16) from an LDS row-major [16][stride] buffer.
// ISA layout: lane<16 -> kb=0, lane>=16 -> kb=8; element j -> K = kb + j (+8 if j>=8).
__device__ __forceinline__ v16bf load_afrag(const bf16_t* a, int stride, int kbase) {
  int l  = threadIdx.x & 31;
  int kb = (l < 16) ? 0 : 8;
  int m  = l & 15;
  union { v16bf v; v8bf h[2]; } u;
  u.h[0] = *(const v8bf*)(a + m * stride + kbase + kb);
  u.h[1] = *(const v8bf*)(a + m * stride + kbase + kb + 16);
  return u.v;
}

// B-fragment: pre-packed tile of 512 bf16; lane l owns 16 contiguous elements.
__device__ __forceinline__ v16bf load_bfrag(const bf16_t* tile) {
  int l = threadIdx.x & 31;
  return *(const v16bf*)(tile + l * 16);
}

__device__ __forceinline__ v8f wmma_bf16(v16bf a, v16bf b, v8f c) {
  return __builtin_amdgcn_wmma_f32_16x16x32_bf16(false, a, false, b, (short)0, c, false, false);
}

// One K-sweep accumulating two output tiles that share the A operand.
// Rolled loop (unroll 1) keeps at most one A- and two B-fragments in flight.
__device__ __forceinline__ void gate_pass(const bf16_t* aLDS, int astride, int KT,
                                          const bf16_t* w0, const bf16_t* w1,
                                          v8f& acc0, v8f& acc1) {
#pragma unroll 1
  for (int kt = 0; kt < KT; ++kt) {
    v16bf af = load_afrag(aLDS, astride, kt * 32);
    acc0 = wmma_bf16(af, load_bfrag(w0 + (size_t)kt * 512), acc0);
    acc1 = wmma_bf16(af, load_bfrag(w1 + (size_t)kt * 512), acc1);
  }
}

// -------- weight packing: f32 [K][N] -> bf16 WMMA B-tiles --------
// LSTM layout: tile = (nh*4 + g)*KT + kt, col = g*HID + nh*16 + (lane&15)
// MLP  layout: tile = nt*KT + kt,         col = nt*16 + (lane&15)
// element within tile = lane*16 + j ; source row kk = kt*32 + (j&7) + 8*(lane>>4) + 16*(j>>3) (kmode remaps)
__global__ void pack_weights(const float* __restrict__ W, bf16_t* __restrict__ dst,
                             int KT, int srcK, int srcN, int nOuter, int lstm, int kmode) {
  int total = (lstm ? nOuter * 4 : nOuter) * KT * 512;
  for (int tid = blockIdx.x * blockDim.x + threadIdx.x; tid < total;
       tid += gridDim.x * blockDim.x) {
    int j    = tid & 15;
    int lane = (tid >> 4) & 31;
    int tile = tid >> 9;
    int kt   = tile % KT;
    int rest = tile / KT;
    int col;
    if (lstm) { int g = rest & 3; int nh = rest >> 2; col = g * HID + nh * 16 + (lane & 15); }
    else      { col = rest * 16 + (lane & 15); }
    int kk = kt * 32 + (j & 7) + 8 * (lane >> 4) + 16 * (j >> 3);
    int k  = (kmode == 1) ? ((kk < 6) ? kk : ((kk >= 32) ? kk - 26 : -1)) : kk;
    float v = 0.0f;
    if (k >= 0 && k < srcK && col < srcN) v = W[(size_t)k * srcN + col];
    dst[tid] = f2bf(v);
  }
}

__global__ void zero_scalar(float* p) { if (threadIdx.x == 0 && blockIdx.x == 0) *p = 0.0f; }

// -------- persistent 2-layer LSTM: 1 workgroup = 16 batch rows, 32 waves --------
// Wave w owns hidden tile nh = w (16 columns) for both layers; c1/c2 live in VGPRs.
__global__ __launch_bounds__(1024, 1) void lstm_kernel(
    const float* __restrict__ seq_in, const int* __restrict__ lengths,
    const float* __restrict__ b1, const float* __restrict__ b2,
    const bf16_t* __restrict__ pw1, const bf16_t* __restrict__ pw2,
    bf16_t* __restrict__ h2out) {
  __shared__ bf16_t a1[16][A1STR];   // [x(6) | zeros(26) | h1(512)]
  __shared__ bf16_t a2[16][A2STR];   // [h1(512) | h2(512)]
  __shared__ int len16[16];

  const int tid = threadIdx.x, bg = blockIdx.x;
  const int nh = tid >> 5;           // wave id == hidden tile 0..31
  const int l = tid & 31, hi = l >> 4, nlane = l & 15;
  const int colb = nh * 16 + nlane;

  for (int i = tid; i < 16 * A1STR; i += 1024) (&a1[0][0])[i] = f2bf(0.0f);
  for (int i = tid; i < 16 * A2STR; i += 1024) (&a2[0][0])[i] = f2bf(0.0f);
  if (tid < 16) len16[tid] = lengths[bg * 16 + tid];
  __syncthreads();

  // t-invariant values hoisted into registers
  float bias1[4], bias2[4];
#pragma unroll
  for (int g = 0; g < 4; ++g) {
    bias1[g] = b1[g * HID + colb];
    bias2[g] = b2[g * HID + colb];
  }
  int lenm[8];
#pragma unroll
  for (int r = 0; r < 8; ++r) lenm[r] = len16[r + hi * 8];

  const bf16_t* w1g[4];
  const bf16_t* w2g[4];
#pragma unroll
  for (int g = 0; g < 4; ++g) {
    w1g[g] = pw1 + (size_t)((nh * 4 + g) * KT1) * 512;
    w2g[g] = pw2 + (size_t)((nh * 4 + g) * KT2) * 512;
  }

  v8f c1, c2;
#pragma unroll
  for (int r = 0; r < 8; ++r) { c1[r] = 0.0f; c2[r] = 0.0f; }

  for (int t = 0; t < TLEN; ++t) {
    if (tid < 96) {  // stage x_t
      int m = tid / 6, c = tid % 6, b = bg * 16 + m;
      a1[m][c] = f2bf(seq_in[((size_t)b * TLEN + t) * DIN + c]);
    }
    __syncthreads();

    // ---- layer 1 gates ----
    v8f acc0, acc1, acc2, acc3;
#pragma unroll
    for (int r = 0; r < 8; ++r) {
      acc0[r] = bias1[0]; acc1[r] = bias1[1];
      acc2[r] = bias1[2]; acc3[r] = bias1[3];
    }
    gate_pass(&a1[0][0], A1STR, KT1, w1g[0], w1g[1], acc0, acc1);
    gate_pass(&a1[0][0], A1STR, KT1, w1g[2], w1g[3], acc2, acc3);
    __syncthreads();  // all waves done reading a1

#pragma unroll
    for (int r = 0; r < 8; ++r) {
      int m = r + hi * 8;
      float iv = sigf(acc0[r]);
      float jv = tanhf(acc1[r]);
      float fv = sigf(acc2[r] + 1.0f);
      float ov = sigf(acc3[r]);
      float cn = c1[r] * fv + iv * jv;
      float hn = tanhf(cn) * ov;
      if (t < lenm[r]) {
        c1[r] = cn;
        bf16_t hb = f2bf(hn);
        a1[m][32 + colb] = hb;   // recurrent input for layer1
        a2[m][colb]      = hb;   // feed-forward input for layer2
      }
    }
    __syncthreads();

    // ---- layer 2 gates ----
#pragma unroll
    for (int r = 0; r < 8; ++r) {
      acc0[r] = bias2[0]; acc1[r] = bias2[1];
      acc2[r] = bias2[2]; acc3[r] = bias2[3];
    }
    gate_pass(&a2[0][0], A2STR, KT2, w2g[0], w2g[1], acc0, acc1);
    gate_pass(&a2[0][0], A2STR, KT2, w2g[2], w2g[3], acc2, acc3);
    __syncthreads();

#pragma unroll
    for (int r = 0; r < 8; ++r) {
      int m = r + hi * 8;
      float iv = sigf(acc0[r]);
      float jv = tanhf(acc1[r]);
      float fv = sigf(acc2[r] + 1.0f);
      float ov = sigf(acc3[r]);
      float cn = c2[r] * fv + iv * jv;
      float hn = tanhf(cn) * ov;
      bool upd = (t < lenm[r]);
      if (upd) { c2[r] = cn; a2[m][HID + colb] = f2bf(hn); }
      h2out[((size_t)(bg * TLEN + t) * 16 + m) * HID + colb] = f2bf(upd ? hn : 0.0f);
    }
    __syncthreads();
  }
}

// one 16x512 @ 512x512 dense layer + ReLU, LDS->LDS (wave w owns n-tiles 2w, 2w+1)
__device__ __forceinline__ void dense512(const bf16_t* xin, bf16_t* xout,
                                         const bf16_t* pw, const float* bias) {
  int tid = threadIdx.x, w = tid >> 5, l = tid & 31, hi = l >> 4, nlane = l & 15;
  v8f acc0, acc1;
  {
    float b0 = bias[(w * 2) * 16 + nlane];
    float b1 = bias[(w * 2 + 1) * 16 + nlane];
#pragma unroll
    for (int r = 0; r < 8; ++r) { acc0[r] = b0; acc1[r] = b1; }
  }
  gate_pass(xin, 512, 32,
            pw + (size_t)((w * 2) * 32) * 512,
            pw + (size_t)((w * 2 + 1) * 32) * 512, acc0, acc1);
#pragma unroll
  for (int r = 0; r < 8; ++r) {
    int m = r + hi * 8;
    xout[m * 512 + (w * 2) * 16 + nlane]     = f2bf(fmaxf(acc0[r], 0.0f));
    xout[m * 512 + (w * 2 + 1) * 16 + nlane] = f2bf(fmaxf(acc1[r], 0.0f));
  }
}

// -------- decoder MLP + masked MSE: 1 workgroup per 16-frame tile --------
__global__ __launch_bounds__(512, 1) void decoder_kernel(
    const bf16_t* __restrict__ h2out, const float* __restrict__ tgt,
    const int* __restrict__ lengths,
    const float* bd1, const float* bd2, const float* bd3, const float* bd4,
    const bf16_t* pd1, const bf16_t* pd2, const bf16_t* pd3, const bf16_t* pd4,
    float* __restrict__ out, float* __restrict__ loss) {
  __shared__ bf16_t xa[16][512];
  __shared__ bf16_t xb[16][512];
  __shared__ bf16_t xc[16][256];
  __shared__ float  outstage[16][4];

  const int tid = threadIdx.x, ft = blockIdx.x;
  const int bg = ft >> 9, t = ft & 511;
  const int w = tid >> 5, l = tid & 31, hi = l >> 4, nlane = l & 15;

  *(v16bf*)(&xa[0][0] + tid * 16) = *(const v16bf*)(h2out + (size_t)ft * 16 * 512 + tid * 16);
  __syncthreads();

  dense512(&xa[0][0], &xb[0][0], pd1, bd1);
  __syncthreads();
  dense512(&xb[0][0], &xa[0][0], pd2, bd2);
  __syncthreads();

  {  // layer 3: 512 -> 256, wave w owns n-tile w
    v8f acc;
    float bv = bd3[w * 16 + nlane];
#pragma unroll
    for (int r = 0; r < 8; ++r) acc[r] = bv;
#pragma unroll 1
    for (int kt = 0; kt < 32; ++kt)
      acc = wmma_bf16(load_afrag(&xa[0][0], 512, kt * 32),
                      load_bfrag(pd3 + (size_t)(w * 32 + kt) * 512), acc);
    int colb = w * 16 + nlane;
#pragma unroll
    for (int r = 0; r < 8; ++r) xc[r + hi * 8][colb] = f2bf(fmaxf(acc[r], 0.0f));
  }
  __syncthreads();

  if (w == 0) {  // layer 4: 256 -> 4 (padded to 16), wave-uniform branch
    v8f acc;
    float bv = (nlane < 4) ? bd4[nlane] : 0.0f;
#pragma unroll
    for (int r = 0; r < 8; ++r) acc[r] = bv;
#pragma unroll 1
    for (int kt = 0; kt < 8; ++kt)
      acc = wmma_bf16(load_afrag(&xc[0][0], 256, kt * 32),
                      load_bfrag(pd4 + (size_t)kt * 512), acc);
    volatile float (*os)[4] = outstage;
    if (nlane < 4) {
#pragma unroll
      for (int r = 0; r < 8; ++r) {
        int m = r + hi * 8, b = bg * 16 + m;
        out[((size_t)b * TLEN + t) * 4 + nlane] = acc[r];
        os[m][nlane] = acc[r];
      }
    }
    if (tid < 16) {  // per-row masked MSE (LDS in-order within wave)
      int m = tid, b = bg * 16 + m;
      const float* tg = tgt + ((size_t)b * TLEN + t) * 4;
      float fm = 0.0f, amax = 0.0f;
#pragma unroll
      for (int k = 0; k < 4; ++k) {
        float d = os[m][k] - tg[k];
        fm += d * d;
        amax = fmaxf(amax, fabsf(tg[k]));
      }
      fm *= 0.25f;
      float msk = (amax > 0.0f) ? 1.0f : 0.0f;
      atomicAdd(loss, fm * msk / ((float)lengths[b] * 64.0f));
    }
  }
}

extern "C" void kernel_launch(void* const* d_in, const int* in_sizes, int n_in,
                              void* d_out, int out_size, void* d_ws, size_t ws_size,
                              hipStream_t stream) {
  (void)in_sizes; (void)n_in; (void)ws_size;
  const float* seq_in  = (const float*)d_in[0];
  const float* seq_tgt = (const float*)d_in[1];
  const int*   seq_len = (const int*)d_in[2];
  const float* W1  = (const float*)d_in[3];
  const float* b1  = (const float*)d_in[4];
  const float* W2  = (const float*)d_in[5];
  const float* b2  = (const float*)d_in[6];
  const float* Wd1 = (const float*)d_in[7];
  const float* bd1 = (const float*)d_in[8];
  const float* Wd2 = (const float*)d_in[9];
  const float* bd2 = (const float*)d_in[10];
  const float* Wd3 = (const float*)d_in[11];
  const float* bd3 = (const float*)d_in[12];
  const float* Wd4 = (const float*)d_in[13];
  const float* bd4 = (const float*)d_in[14];

  char* ws = (char*)d_ws;
  size_t off = 0;
  bf16_t* pw1 = (bf16_t*)(ws + off); off += (size_t)32 * 4 * KT1 * 512 * 2;
  bf16_t* pw2 = (bf16_t*)(ws + off); off += (size_t)32 * 4 * KT2 * 512 * 2;
  bf16_t* pd1 = (bf16_t*)(ws + off); off += (size_t)32 * 32 * 512 * 2;
  bf16_t* pd2 = (bf16_t*)(ws + off); off += (size_t)32 * 32 * 512 * 2;
  bf16_t* pd3 = (bf16_t*)(ws + off); off += (size_t)16 * 32 * 512 * 2;
  bf16_t* pd4 = (bf16_t*)(ws + off); off += (size_t)8 * 512 * 2;
  bf16_t* h2o = (bf16_t*)(ws + off);

  auto nb = [](size_t n) { return (int)((n + 255) / 256); };
  pack_weights<<<nb((size_t)32*4*KT1*512), 256, 0, stream>>>(W1,  pw1, KT1, 518, 2048, 32, 1, 1);
  pack_weights<<<nb((size_t)32*4*KT2*512), 256, 0, stream>>>(W2,  pw2, KT2, 1024, 2048, 32, 1, 0);
  pack_weights<<<nb((size_t)32*32*512),    256, 0, stream>>>(Wd1, pd1, 32, 512, 512, 32, 0, 0);
  pack_weights<<<nb((size_t)32*32*512),    256, 0, stream>>>(Wd2, pd2, 32, 512, 512, 32, 0, 0);
  pack_weights<<<nb((size_t)16*32*512),    256, 0, stream>>>(Wd3, pd3, 32, 512, 256, 16, 0, 0);
  pack_weights<<<nb((size_t)8*512),        256, 0, stream>>>(Wd4, pd4,  8, 256,   4,  1, 0, 0);

  float* outp = (float*)d_out;
  float* lossp = outp + (out_size - 1);
  zero_scalar<<<1, 32, 0, stream>>>(lossp);

  lstm_kernel<<<4, 1024, 0, stream>>>(seq_in, seq_len, b1, b2, pw1, pw2, h2o);
  decoder_kernel<<<2048, 512, 0, stream>>>(h2o, seq_tgt, seq_len,
                                           bd1, bd2, bd3, bd4,
                                           pd1, pd2, pd3, pd4, outp, lossp);
}
